// AttentionBlock_5446018532008
// MI455X (gfx1250) — matrix operand order repaired and checked by reading, not compiled, not run on hardware
//
#include <hip/hip_runtime.h>

// ---------------------------------------------------------------------------
// AttentionBlock for MI455X (gfx1250, wave32, WMMA 16x16x32 f16, f32 accum)
// B=32, C=256, S=1024, heads=4, d_head=64
// v2: LDS-shared K/V blocks (4 q-tiles/block), double-buffered async-to-LDS
//     staging (global_load_async_to_lds_b128 + s_wait_asynccnt), and 16x64
//     output tiles in both projection GEMMs (A-fragment reuse).
// ---------------------------------------------------------------------------

typedef __attribute__((ext_vector_type(16))) _Float16 v16h;
typedef __attribute__((ext_vector_type(8)))  float    v8f;

#define NB   32
#define NC   256
#define NS   1024
#define NH   4
#define ND   64
#define NQKV 768

union H2 { unsigned int u; _Float16 h[2]; };

// A-fragment (16x32 f16). Lane L: row m = L&15, half hi = L>>4.
// VGPR i: K = 8*hi+2i (i<4) or 16+8*hi+2*(i-4). Row-major, K contiguous.
__device__ __forceinline__ v16h load_a_f16(const _Float16* p, int stride,
                                           int k0, int lane) {
  int m = lane & 15, hi = lane >> 4;
  const _Float16* row = p + m * stride + k0;
  v16h a;
#pragma unroll
  for (int i = 0; i < 8; ++i) {
    int k = (i < 4) ? (8 * hi + 2 * i) : (16 + 8 * hi + 2 * (i - 4));
    H2 t; t.u = *reinterpret_cast<const unsigned int*>(row + k);
    a[2 * i] = t.h[0]; a[2 * i + 1] = t.h[1];
  }
  return a;
}

// B-fragment (32x16 f16, KxN). Lane L: col n = L&15, half hi = L>>4.
// VGPR v: K = 8*hi+2v (v<4) or 16+8*hi+2*(v-4). N-major (stride halves/column), K contiguous.
__device__ __forceinline__ v16h load_b_f16(const _Float16* p, int stride,
                                           int k0, int lane) {
  int n = lane & 15, hi = lane >> 4;
  const _Float16* row = p + n * stride + k0;
  v16h b;
#pragma unroll
  for (int v = 0; v < 8; ++v) {
    int k = (v < 4) ? (8 * hi + 2 * v) : (16 + 8 * hi + 2 * (v - 4));
    H2 t; t.u = *reinterpret_cast<const unsigned int*>(row + k);
    b[2 * v] = t.h[0]; b[2 * v + 1] = t.h[1];
  }
  return b;
}

__device__ __forceinline__ v8f wmma16(v16h a, v16h b, v8f c) {
  return __builtin_amdgcn_wmma_f32_16x16x32_f16(false, a, false, b,
                                                (short)0, c, false, false);
}

// Async 16B copy global -> LDS (per-lane addresses, ASYNCcnt-tracked).
// Low 32 bits of a generic LDS pointer are the LDS byte address (ISA §10.2).
__device__ __forceinline__ void async_copy16(const _Float16* g, _Float16* l) {
  unsigned lds = (unsigned)(uintptr_t)l;
  unsigned long long ga = (unsigned long long)(uintptr_t)g;
  asm volatile("global_load_async_to_lds_b128 %0, %1, off"
               :: "v"(lds), "v"(ga) : "memory");
}
__device__ __forceinline__ void wait_async0() {
  asm volatile("s_wait_asynccnt 0" ::: "memory");
}

// ---------- Stage 1a: x [B,C,S] f32 -> xs [B,S,C] f16 ----------------------
__global__ void k_convert_x(const float* __restrict__ x, _Float16* __restrict__ xs) {
  long i = (long)blockIdx.x * blockDim.x + threadIdx.x;   // over B*S*C
  int c = (int)(i & (NC - 1));
  long t = i >> 8;
  int s = (int)(t & (NS - 1));
  int b = (int)(t >> 10);
  xs[i] = (_Float16)x[((long)b * NC + c) * NS + s];
}

// ---------- Stage 1b: transpose weights to [N][K] f16 ----------------------
__global__ void k_convert_w(const float* __restrict__ Wqkv,
                            const float* __restrict__ Wout,
                            _Float16* __restrict__ Wqkvt,
                            _Float16* __restrict__ Woutt) {
  int i = blockIdx.x * blockDim.x + threadIdx.x;
  if (i < NQKV * NC) {
    int n = i / NC, k = i % NC;
    Wqkvt[i] = (_Float16)Wqkv[k * NQKV + n];
  } else {
    int j = i - NQKV * NC;
    int n = j / NC, k = j % NC;
    Woutt[j] = (_Float16)Wout[k * NC + n];
  }
}

// ---------- Stage 2: QKV GEMM, one wave = 16x64 output tile ----------------
__global__ __launch_bounds__(128)
void k_qkv_gemm(const _Float16* __restrict__ xs, const _Float16* __restrict__ wt,
                const float* __restrict__ bias, _Float16* __restrict__ Q,
                _Float16* __restrict__ K, _Float16* __restrict__ Vt) {
  int wave = (blockIdx.x * blockDim.x + threadIdx.x) >> 5;
  int lane = threadIdx.x & 31;
  int ng = wave % 12; int t = wave / 12;          // 12 groups of 4 n-tiles
  int stile = t & 63; int b = t >> 6;
  const _Float16* A = xs + ((long)b * NS + stile * 16) * NC;
  v8f acc[4] = {{}, {}, {}, {}};
#pragma unroll
  for (int k0 = 0; k0 < NC; k0 += 32) {
    v16h a = load_a_f16(A, NC, k0, lane);         // loaded once, used 4x
#pragma unroll
    for (int nt = 0; nt < 4; ++nt)
      acc[nt] = wmma16(a, load_b_f16(wt + (long)((ng * 4 + nt) * 16) * NC,
                                     NC, k0, lane), acc[nt]);
  }
  int n = lane & 15, hi = lane >> 4;
#pragma unroll
  for (int nt = 0; nt < 4; ++nt) {
    int col = (ng * 4 + nt) * 16 + n;
    int h = col / 192, r = col % 192, which = r / 64, d = r & 63;
    float bval = bias[col];
#pragma unroll
    for (int rr = 0; rr < 8; ++rr) {
      int s = stile * 16 + rr + 8 * hi;
      float v = acc[nt][rr] + bval;
      if (which == 0)      Q [((long)(b * NH + h) * NS + s) * ND + d] = (_Float16)v;
      else if (which == 1) K [((long)(b * NH + h) * NS + s) * ND + d] = (_Float16)v;
      else                 Vt[((long)(b * NH + h) * ND + d) * NS + s] = (_Float16)v;
    }
  }
}

// ---------- Stage 3: fused flash attention ---------------------------------
// Block = 4 waves = 4 consecutive q-tiles of one (b,h); each 32-key block of
// K/V is async-staged into double-buffered LDS once and shared by all waves.
__global__ __launch_bounds__(128)
void k_attn(const _Float16* __restrict__ Q, const _Float16* __restrict__ K,
            const _Float16* __restrict__ Vt, _Float16* __restrict__ AO) {
  __shared__ __align__(16) _Float16 sK[2][32][72];   // [key][d], padded rows
  __shared__ __align__(16) _Float16 sV[2][64][40];   // [d][key], padded rows
  __shared__ float sP[4][16][33];                    // per-wave P tile
  int tid = threadIdx.x;
  int wid = tid >> 5, lane = tid & 31;
  int qgrp = blockIdx.x & 15; int bh = blockIdx.x >> 4;
  int b = bh >> 2, h = bh & 3;
  int qtile = qgrp * 4 + wid;
  const _Float16* Qb = Q  + ((long)bh * NS + qtile * 16) * ND;
  const _Float16* Kb = K  + (long)bh * NS * ND;
  const _Float16* Vb = Vt + (long)bh * ND * NS;
  int n = lane & 15, hi = lane >> 4;

  // async-stage one 32-key block (4KB K + 4KB V) into LDS buffer `buf`
  auto stage = [&](int buf, int jb) {
#pragma unroll
    for (int c2 = 0; c2 < 2; ++c2) {
      int c = tid + c2 * 128;                      // 256 16B chunks each
      async_copy16(Kb + (long)(jb + (c >> 3)) * ND + (c & 7) * 8,
                   &sK[buf][c >> 3][(c & 7) * 8]);
      async_copy16(Vb + (long)(c >> 2) * NS + jb + (c & 3) * 8,
                   &sV[buf][c >> 2][(c & 3) * 8]);
    }
  };

  stage(0, 0);                                     // prologue fill
  v16h aq0 = load_a_f16(Qb, ND, 0,  lane);
  v16h aq1 = load_a_f16(Qb, ND, 32, lane);

  float mrow[8], lrow[8];
  v8f o0 = {}, o1 = {}, o2 = {}, o3 = {};
#pragma unroll
  for (int r = 0; r < 8; ++r) { mrow[r] = -1e30f; lrow[r] = 0.f; }
  const float scale = 0.125f;                      // 64^-0.5

  wait_async0();
  __syncthreads();

  for (int jb = 0; jb < NS; jb += 32) {
    int cur = (jb >> 5) & 1;
    if (jb + 32 < NS) stage(cur ^ 1, jb + 32);     // pipelined prefetch

    // scores S[16,32] = Q(16x64) * K^T : 4 WMMAs, B-frags from LDS
    v8f s0 = {}, s1 = {};
    s0 = wmma16(aq0, load_b_f16(&sK[cur][0][0],  72, 0,  lane), s0);
    s0 = wmma16(aq1, load_b_f16(&sK[cur][0][0],  72, 32, lane), s0);
    s1 = wmma16(aq0, load_b_f16(&sK[cur][16][0], 72, 0,  lane), s1);
    s1 = wmma16(aq1, load_b_f16(&sK[cur][16][0], 72, 32, lane), s1);

    // online softmax across this block's 32 keys (16-lane row reductions)
#pragma unroll
    for (int r = 0; r < 8; ++r) {
      float a0 = s0[r] * scale, a1 = s1[r] * scale;
      float mx = fmaxf(a0, a1);
#pragma unroll
      for (int d = 1; d < 16; d <<= 1) mx = fmaxf(mx, __shfl_xor(mx, d, 32));
      float mnew  = fmaxf(mrow[r], mx);
      float alpha = __expf(mrow[r] - mnew);
      float p0 = __expf(a0 - mnew), p1 = __expf(a1 - mnew);
      float sum = p0 + p1;
#pragma unroll
      for (int d = 1; d < 16; d <<= 1) sum += __shfl_xor(sum, d, 32);
      lrow[r] = lrow[r] * alpha + sum;
      mrow[r] = mnew;
      o0[r] *= alpha; o1[r] *= alpha; o2[r] *= alpha; o3[r] *= alpha;
      sP[wid][r + 8 * hi][n]      = p0;            // C-frag -> LDS transpose
      sP[wid][r + 8 * hi][16 + n] = p1;
    }
    __syncthreads();                               // sP visible to own wave

    v16h ap;                                       // P as A-fragment
#pragma unroll
    for (int i = 0; i < 8; ++i) {
      int k = (i < 4) ? (8 * hi + 2 * i) : (16 + 8 * hi + 2 * (i - 4));
      ap[2 * i]     = (_Float16)sP[wid][lane & 15][k];
      ap[2 * i + 1] = (_Float16)sP[wid][lane & 15][k + 1];
    }

    // O[16,64] += P(16x32) * V(32x64), B-frags from LDS (V is [d][key])
    o0 = wmma16(ap, load_b_f16(&sV[cur][ 0][0], 40, 0, lane), o0);
    o1 = wmma16(ap, load_b_f16(&sV[cur][16][0], 40, 0, lane), o1);
    o2 = wmma16(ap, load_b_f16(&sV[cur][32][0], 40, 0, lane), o2);
    o3 = wmma16(ap, load_b_f16(&sV[cur][48][0], 40, 0, lane), o3);

    wait_async0();                                 // next buffer landed
    __syncthreads();                               // all waves done with cur
  }

  // normalize and write AO [B,S,256] f16 (c = h*64 + d)
#pragma unroll
  for (int r = 0; r < 8; ++r) {
    float inv = 1.0f / (lrow[r] > 0.f ? lrow[r] : 1.f);
    int s = qtile * 16 + r + 8 * hi;
    long rowoff = ((long)b * NS + s) * NC + h * ND;
    AO[rowoff + 0 * 16 + n] = (_Float16)(o0[r] * inv);
    AO[rowoff + 1 * 16 + n] = (_Float16)(o1[r] * inv);
    AO[rowoff + 2 * 16 + n] = (_Float16)(o2[r] * inv);
    AO[rowoff + 3 * 16 + n] = (_Float16)(o3[r] * inv);
  }
}

// ---------- Stage 4: out = AO @ W_out + b_out + x, one wave = 16x64 --------
__global__ __launch_bounds__(128)
void k_out_gemm(const _Float16* __restrict__ AO, const _Float16* __restrict__ Wt,
                const float* __restrict__ bias, const float* __restrict__ x,
                float* __restrict__ out) {
  int wave = (blockIdx.x * blockDim.x + threadIdx.x) >> 5;
  int lane = threadIdx.x & 31;
  int ng = wave & 3;  int t = wave >> 2;           // 4 groups of 4 n-tiles
  int stile = t & 63; int b = t >> 6;
  const _Float16* A = AO + ((long)b * NS + stile * 16) * NC;
  v8f acc[4] = {{}, {}, {}, {}};
#pragma unroll
  for (int k0 = 0; k0 < NC; k0 += 32) {
    v16h a = load_a_f16(A, NC, k0, lane);
#pragma unroll
    for (int nt = 0; nt < 4; ++nt)
      acc[nt] = wmma16(a, load_b_f16(Wt + (long)((ng * 4 + nt) * 16) * NC,
                                     NC, k0, lane), acc[nt]);
  }
  int n = lane & 15, hi = lane >> 4;
#pragma unroll
  for (int nt = 0; nt < 4; ++nt) {
    int c = (ng * 4 + nt) * 16 + n;
    float bval = bias[c];
#pragma unroll
    for (int r = 0; r < 8; ++r) {
      int s = stile * 16 + r + 8 * hi;
      long idx = ((long)b * NC + c) * NS + s;
      out[idx] = acc[nt][r] + bval + x[idx];       // residual straight from x
    }
  }
}

// ---------------------------------------------------------------------------
extern "C" void kernel_launch(void* const* d_in, const int* in_sizes, int n_in,
                              void* d_out, int out_size, void* d_ws, size_t ws_size,
                              hipStream_t stream) {
  const float* x     = (const float*)d_in[0];
  const float* W_qkv = (const float*)d_in[1];
  const float* b_qkv = (const float*)d_in[2];
  const float* W_out = (const float*)d_in[3];
  const float* b_out = (const float*)d_in[4];
  float* out = (float*)d_out;

  char* w = (char*)d_ws;
  size_t off = 0;
  auto alloc = [&](size_t bytes) -> void* {
    void* p = w + off;
    off = (off + bytes + 255) & ~(size_t)255;
    return p;
  };
  const size_t nBSC = (size_t)NB * NS * NC;         // 8,388,608
  _Float16* xs    = (_Float16*)alloc(nBSC * 2);
  _Float16* Wqkvt = (_Float16*)alloc((size_t)NQKV * NC * 2);
  _Float16* Woutt = (_Float16*)alloc((size_t)NC * NC * 2);
  _Float16* Qa    = (_Float16*)alloc(nBSC * 2);     // [B,H,S,D]
  _Float16* Ka    = (_Float16*)alloc(nBSC * 2);     // [B,H,S,D]
  _Float16* Vta   = (_Float16*)alloc(nBSC * 2);     // [B,H,D,S]
  _Float16* AO    = (_Float16*)alloc(nBSC * 2);     // [B,S,C]
  (void)ws_size; (void)in_sizes; (void)n_in; (void)out_size;

  // Stage 1
  k_convert_x<<<(int)(nBSC / 256), 256, 0, stream>>>(x, xs);
  k_convert_w<<<(NQKV * NC + NC * NC) / 256, 256, 0, stream>>>(W_qkv, W_out,
                                                               Wqkvt, Woutt);
  // Stage 2: 32*64*12 = 24576 waves, 4 waves/block
  k_qkv_gemm<<<24576 / 4, 128, 0, stream>>>(xs, Wqkvt, b_qkv, Qa, Ka, Vta);
  // Stage 3: 32*4*16 = 2048 blocks of 4 waves
  k_attn<<<2048, 128, 0, stream>>>(Qa, Ka, Vta, AO);
  // Stage 4: 32*64*4 = 8192 waves, 4 waves/block
  k_out_gemm<<<8192 / 4, 128, 0, stream>>>(AO, Woutt, b_out, x, out);
}
